// BiLSTM_CRF_40596030882307
// MI455X (gfx1250) — matrix operand order repaired and checked by reading, not compiled, not run on hardware
//
#include <hip/hip_runtime.h>
#include <stdint.h>

typedef __attribute__((ext_vector_type(16))) __bf16 v16bf;
typedef __attribute__((ext_vector_type(8)))  float  v8f;
typedef unsigned int u32x4 __attribute__((ext_vector_type(4)));
typedef unsigned int u32x8 __attribute__((ext_vector_type(8)));

union BF16x16 {
    v16bf v;
    uint4 q[2];
};

__device__ __forceinline__ unsigned short f2bf(float f) {
    union { float f; unsigned int u; } x; x.f = f;
    unsigned int u = x.u + 0x7FFFu + ((x.u >> 16) & 1u);   // round-to-nearest-even
    return (unsigned short)(u >> 16);
}
__device__ __forceinline__ float sigm(float x) { return 1.0f / (1.0f + __expf(-x)); }

#define L_SEQ 8192
#define EDIM  256
#define H4    1024
#define KTAG  10
#define START_TAG 8
#define STOP_TAG  9

// ---------------------------------------------------------------------------
// TDM: async DMA of one 4KB row (1024 f32) from global memory into LDS.
// D# per CDNA5 ISA: group0 {count,lds_addr,global_addr,type=2},
// group1 {mask=0, data_size=4B, tensor_dim0=1024, tensor_dim1=8192,
//         tile_dim0=1024, tile_dim1=1, tensor_dim0_stride=1024}.
// Tracked by TENSORcnt; completion enforced with s_wait_tensorcnt.
// ---------------------------------------------------------------------------
__device__ __forceinline__ void tdm_load_row(const float* src, unsigned lds_off)
{
    union { unsigned int w[4]; u32x4 v; } g0;
    union { unsigned int w[8]; u32x8 v; } g1;
    unsigned long long ga = (unsigned long long)(size_t)src;
    g0.w[0] = 1u;                                              // count=1, user D#
    g0.w[1] = lds_off;                                         // lds_addr [63:32]
    g0.w[2] = (unsigned)(ga & 0xFFFFFFFFu);                    // global_addr lo
    g0.w[3] = (unsigned)((ga >> 32) & 0x01FFFFFFu) | (2u << 30); // addr hi | type=2
    g1.w[0] = (2u << 16);                                      // data_size=4B
    g1.w[1] = (1024u & 0xFFFFu) << 16;                         // tensor_dim0 lo16
    g1.w[2] = (1024u >> 16) | ((8192u & 0xFFFFu) << 16);       // dim0 hi | dim1 lo
    g1.w[3] = (8192u >> 16) | (1024u << 16);                   // dim1 hi | tile_dim0
    g1.w[4] = 1u;                                              // tile_dim1=1
    g1.w[5] = 1024u;                                           // dim0_stride lo32
    g1.w[6] = 0u;
    g1.w[7] = 0u;
    asm volatile("tensor_load_to_lds %0, %1" :: "s"(g0.v), "s"(g1.v) : "memory");
}

// ---------------------------------------------------------------------------
// K0: convert the four [1024,256] f32 weight matrices to bf16
// ---------------------------------------------------------------------------
__global__ void k_w2bf(const float* __restrict__ s0, const float* __restrict__ s1,
                       const float* __restrict__ s2, const float* __restrict__ s3,
                       unsigned short* __restrict__ d0, unsigned short* __restrict__ d1,
                       unsigned short* __restrict__ d2, unsigned short* __restrict__ d3, int n)
{
    int i = blockIdx.x * blockDim.x + threadIdx.x;
    if (i >= n) return;
    const float* s; unsigned short* d;
    switch (blockIdx.y) {
        case 0:  s = s0; d = d0; break;
        case 1:  s = s1; d = d1; break;
        case 2:  s = s2; d = d2; break;
        default: s = s3; d = d3; break;
    }
    d[i] = f2bf(s[i]);
}

// ---------------------------------------------------------------------------
// K1: embedding gather + f32->bf16   x[t,e] = emb[sentence[t], e]
// ---------------------------------------------------------------------------
__global__ void k_embed(const int* __restrict__ sent, const float* __restrict__ emb,
                        unsigned short* __restrict__ xbf)
{
    int t  = blockIdx.x;            // 8192
    int e4 = threadIdx.x;           // 64 threads * 4 elems
    float4 v = *(const float4*)(emb + (size_t)sent[t] * EDIM + e4 * 4);
    ushort4 o;
    o.x = f2bf(v.x); o.y = f2bf(v.y); o.z = f2bf(v.z); o.w = f2bf(v.w);
    *(ushort4*)(xbf + (size_t)t * EDIM + e4 * 4) = o;
}

// ---------------------------------------------------------------------------
// K2: input projection GEMM via WMMA bf16:
//     g[t,n] = sum_k x[t,k]*Wih[n,k] + bih[n] + bhh[n]     (M=8192,N=1024,K=256)
// ---------------------------------------------------------------------------
__global__ void __launch_bounds__(32) k_inproj(
    const unsigned short* __restrict__ xbf,
    const unsigned short* __restrict__ wf, const unsigned short* __restrict__ wb,
    const float* __restrict__ bihf, const float* __restrict__ bhhf,
    const float* __restrict__ bihb, const float* __restrict__ bhhb,
    float* __restrict__ gf, float* __restrict__ gb)
{
    const int lane = threadIdx.x;
    const int m0   = blockIdx.x * 16;
    const int n0   = blockIdx.y * 16;
    const int dir  = blockIdx.z;
    const unsigned short* w = dir ? wb : wf;
    const float* bi = dir ? bihb : bihf;
    const float* bh = dir ? bhhb : bhhf;
    float* g        = dir ? gb : gf;

    const int half = lane >> 4;
    const int l15  = lane & 15;
    const int trow = m0 + l15;
    const int nrow = n0 + l15;

    v8f acc = {};
#pragma unroll
    for (int kc = 0; kc < 8; ++kc) {
        BF16x16 A, B;
        int abase = kc * 32 + half * 16;
        A.q[0] = *(const uint4*)(xbf + (size_t)trow * EDIM + abase);
        A.q[1] = *(const uint4*)(xbf + (size_t)trow * EDIM + abase + 8);
        int bbase = kc * 32 + half * 16;
        const uint4* bp = (const uint4*)(w + (size_t)nrow * EDIM + bbase);
        B.q[0] = bp[0]; B.q[1] = bp[1];
        acc = __builtin_amdgcn_wmma_f32_16x16x32_bf16(false, A.v, false, B.v,
                                                      (short)0, acc, false, false);
    }
#pragma unroll
    for (int r = 0; r < 8; ++r) {
        int m = m0 + r + half * 8;
        int n = n0 + l15;
        g[(size_t)m * H4 + n] = acc[r] + bi[n] + bh[n];
    }
}

// ---------------------------------------------------------------------------
// K3: sequential LSTM recurrence, one 32-wave workgroup per direction.
// Whh (bf16, 512KB) register-resident (128 VGPRs/wave of B-tiles).
// Per step: 16 WMMAs/wave; the next step's 4KB gate-preactivation row is
// TDM-prefetched into LDS (double-buffered) while WMMAs run.
// ---------------------------------------------------------------------------
__global__ void __launch_bounds__(1024) k_lstm(
    const float* __restrict__ gf, const float* __restrict__ gb,
    const unsigned short* __restrict__ whhf, const unsigned short* __restrict__ whhb,
    float* __restrict__ hf, float* __restrict__ hb)
{
    __shared__ float gbuf[H4];
    __shared__ __align__(16) float gpre[2][H4];      // TDM double buffer
    __shared__ __align__(16) unsigned short hbf[256];

    const int dir = blockIdx.x;
    const float* gin          = dir ? gb   : gf;
    const unsigned short* whh = dir ? whhb : whhf;
    float* hout               = dir ? hb   : hf;

    const int tid  = threadIdx.x;
    const int wv   = tid >> 5;      // 0..31
    const int lane = tid & 31;
    const int half = lane >> 4;
    const int l15  = lane & 15;

    // preload weights: wave wv owns outputs [32*wv, 32*wv+32)
    BF16x16 Breg[2][8];
#pragma unroll
    for (int tile = 0; tile < 2; ++tile) {
        int n = wv * 32 + tile * 16 + l15;
#pragma unroll
        for (int kc = 0; kc < 8; ++kc) {
            int kb = kc * 32 + half * 16;
            const uint4* p = (const uint4*)(whh + (size_t)n * 256 + kb);
            Breg[tile][kc].q[0] = p[0];
            Breg[tile][kc].q[1] = p[1];
        }
    }

    const unsigned gpre_lds = (unsigned)(size_t)(&gpre[0][0]);

    // TDM prefetch of the first row
    if (wv == 0) {
        int t0 = dir ? (L_SEQ - 1) : 0;
        tdm_load_row(gin + (size_t)t0 * H4, gpre_lds);
    }

    if (tid < 256) hbf[tid] = 0;    // h0 = 0
    float c = 0.0f;                 // cell state lives in a register
    int buf = 0;
    __syncthreads();

    for (int s = 0; s < L_SEQ; ++s) {
        const int t = dir ? (L_SEQ - 1 - s) : s;

        v8f acc0 = {}, acc1 = {};
#pragma unroll
        for (int kc = 0; kc < 8; ++kc) {
            BF16x16 A;                       // all 16 rows = h (LDS broadcast reads)
            int base = kc * 32 + half * 16;
            A.q[0] = *(const uint4*)(hbf + base);
            A.q[1] = *(const uint4*)(hbf + base + 8);
            acc0 = __builtin_amdgcn_wmma_f32_16x16x32_bf16(false, A.v, false, Breg[0][kc].v,
                                                           (short)0, acc0, false, false);
            acc1 = __builtin_amdgcn_wmma_f32_16x16x32_bf16(false, A.v, false, Breg[1][kc].v,
                                                           (short)0, acc1, false, false);
        }

        // overlap: wave0 launches the TDM for step s+1, then ensures row s landed
        if (wv == 0) {
            int sn = (s + 1 < L_SEQ) ? (s + 1) : s;
            int tn = dir ? (L_SEQ - 1 - sn) : sn;
            tdm_load_row(gin + (size_t)tn * H4, gpre_lds + (unsigned)((buf ^ 1) * (H4 * 4)));
            __builtin_amdgcn_s_wait_tensorcnt(1);   // oldest (row s) complete
        }
        __syncthreads();                            // gpre[buf] visible to all waves

        if (lane < 16) {                     // D row 0: lanes 0-15, component 0
            int n0 = wv * 32 + lane;
            gbuf[n0]      = acc0[0] + gpre[buf][n0];
            gbuf[n0 + 16] = acc1[0] + gpre[buf][n0 + 16];
        }
        __syncthreads();
        if (tid < 256) {                     // gate order i,f,g,o (torch)
            float ig = sigm(gbuf[tid]);
            float fg = sigm(gbuf[256 + tid]);
            float gg = tanhf(gbuf[512 + tid]);
            float og = sigm(gbuf[768 + tid]);
            c = fg * c + ig * gg;
            float h = og * tanhf(c);
            hout[(size_t)t * 256 + tid] = h;
            hbf[tid] = f2bf(h);
        }
        __syncthreads();
        buf ^= 1;
    }
}

// ---------------------------------------------------------------------------
// K4: feats[t,k] = [hf|hb][t,:] . W_tag[k,:] + b_tag[k]   (K=10, padded to 16)
// ---------------------------------------------------------------------------
__global__ void k_feats(const float* __restrict__ hf, const float* __restrict__ hb,
                        const float* __restrict__ Wtag, const float* __restrict__ btag,
                        float* __restrict__ feats)
{
    int gid = blockIdx.x * blockDim.x + threadIdx.x;
    int t = gid >> 4, k = gid & 15;
    if (t >= L_SEQ) return;
    if (k < KTAG) {
        const float* w = Wtag + k * 512;
        const float* a = hf + (size_t)t * 256;
        const float* b = hb + (size_t)t * 256;
        float s = btag[k];
#pragma unroll 4
        for (int j = 0; j < 256; ++j) s += a[j] * w[j];
#pragma unroll 4
        for (int j = 0; j < 256; ++j) s += b[j] * w[256 + j];
        feats[t * 16 + k] = s;
    } else {
        feats[t * 16 + k] = 0.0f;
    }
}

// ---------------------------------------------------------------------------
// K5: Viterbi forward (single wave, shfl max-plus) + LDS-packed backpointers
//     + serial traceback.  out[0]=score, out[1..8192]=path (as float)
// ---------------------------------------------------------------------------
__global__ void __launch_bounds__(32) k_viterbi(const float* __restrict__ feats,
                                                const float* __restrict__ trans,
                                                float* __restrict__ out)
{
    __shared__ unsigned int  bplo[L_SEQ];   // tags 0..7, 4 bits each
    __shared__ unsigned char bphi[L_SEQ];   // tags 8..9
    __shared__ float ltr[KTAG * KTAG];

    const int lane = threadIdx.x;
    for (int i = lane; i < KTAG * KTAG; i += 32) ltr[i] = trans[i];
    __syncthreads();

    float v = (lane == START_TAG) ? 0.0f : -10000.0f;

    for (int t = 0; t < L_SEQ; ++t) {
        float best = -3.4e38f; int bi = 0;
#pragma unroll
        for (int p = 0; p < KTAG; ++p) {
            float vp = __shfl(v, p, 32);
            float sc = vp + ((lane < KTAG) ? ltr[lane * KTAG + p] : -3.4e38f);
            if (sc > best) { best = sc; bi = p; }
        }
        float fv = (lane < KTAG) ? feats[t * 16 + lane] : 0.0f;
        v = best + fv;

        unsigned int lo = 0;
#pragma unroll
        for (int p = 0; p < 8; ++p)
            lo |= ((unsigned)__shfl(bi, p, 32) & 15u) << (4 * p);
        unsigned int hi = (__shfl(bi, 8, 32) & 15) | ((__shfl(bi, 9, 32) & 15) << 4);
        if (lane == 0) { bplo[t] = lo; bphi[t] = (unsigned char)hi; }
    }

    float term = v + ((lane < KTAG) ? ltr[STOP_TAG * KTAG + lane] : -3.4e38f);
    float bscore = -3.4e38f; int last = 0;
#pragma unroll
    for (int p = 0; p < KTAG; ++p) {
        float tp = __shfl(term, p, 32);
        if (tp > bscore) { bscore = tp; last = p; }
    }
    if (lane == 0) {
        out[0] = bscore;
        int tag = last;
        out[1 + (L_SEQ - 1)] = (float)tag;
        for (int t = L_SEQ - 1; t >= 1; --t) {
            unsigned int lo = bplo[t]; unsigned int hi = bphi[t];
            tag = (tag < 8) ? (int)((lo >> (4 * tag)) & 15u)
                            : (int)((hi >> (4 * (tag - 8))) & 15u);
            out[t] = (float)tag;       // position t-1 lands at out[1+(t-1)]
        }
    }
}

// ---------------------------------------------------------------------------
extern "C" void kernel_launch(void* const* d_in, const int* in_sizes, int n_in,
                              void* d_out, int out_size, void* d_ws, size_t ws_size,
                              hipStream_t stream)
{
    const int*   sent   = (const int*)  d_in[0];
    const float* emb    = (const float*)d_in[1];
    const float* Wih_f  = (const float*)d_in[2];
    const float* Whh_f  = (const float*)d_in[3];
    const float* bih_f  = (const float*)d_in[4];
    const float* bhh_f  = (const float*)d_in[5];
    const float* Wih_b  = (const float*)d_in[6];
    const float* Whh_b  = (const float*)d_in[7];
    const float* bih_b  = (const float*)d_in[8];
    const float* bhh_b  = (const float*)d_in[9];
    const float* W_tag  = (const float*)d_in[10];
    const float* b_tag  = (const float*)d_in[11];
    const float* transi = (const float*)d_in[12];
    float* out = (float*)d_out;

    // workspace carve-up (256B aligned slices)
    char* p = (char*)d_ws;
    auto carve = [&](size_t bytes) { char* r = p; p += (bytes + 255) & ~(size_t)255; return (void*)r; };
    unsigned short* xbf   = (unsigned short*)carve((size_t)L_SEQ * EDIM * 2);
    unsigned short* wihf  = (unsigned short*)carve((size_t)H4 * EDIM * 2);
    unsigned short* whhf  = (unsigned short*)carve((size_t)H4 * EDIM * 2);
    unsigned short* wihb  = (unsigned short*)carve((size_t)H4 * EDIM * 2);
    unsigned short* whhb  = (unsigned short*)carve((size_t)H4 * EDIM * 2);
    float* gf    = (float*)carve((size_t)L_SEQ * H4 * 4);
    float* gb    = (float*)carve((size_t)L_SEQ * H4 * 4);
    float* hfbuf = (float*)carve((size_t)L_SEQ * 256 * 4);
    float* hbbuf = (float*)carve((size_t)L_SEQ * 256 * 4);
    float* feats = (float*)carve((size_t)L_SEQ * 16 * 4);

    // K0: weights -> bf16
    {
        int n = H4 * EDIM;                         // 262144
        dim3 g((n + 255) / 256, 4);
        k_w2bf<<<g, 256, 0, stream>>>(Wih_f, Whh_f, Wih_b, Whh_b,
                                      wihf, whhf, wihb, whhb, n);
    }
    // K1: embedding gather
    k_embed<<<dim3(L_SEQ), dim3(64), 0, stream>>>(sent, emb, xbf);

    // K2: input projections (both directions), WMMA
    k_inproj<<<dim3(L_SEQ / 16, H4 / 16, 2), dim3(32), 0, stream>>>(
        xbf, wihf, wihb, bih_f, bhh_f, bih_b, bhh_b, gf, gb);

    // K3: sequential BiLSTM recurrence (2 workgroups: fwd + bwd), TDM-prefetched
    k_lstm<<<dim3(2), dim3(1024), 0, stream>>>(gf, gb, whhf, whhb, hfbuf, hbbuf);

    // K4: tag-space projection
    k_feats<<<dim3((L_SEQ * 16) / 256), dim3(256), 0, stream>>>(hfbuf, hbbuf, W_tag, b_tag, feats);

    // K5: Viterbi + traceback
    k_viterbi<<<dim3(1), dim3(32), 0, stream>>>(feats, transi, out);
}
